// MultiScaleAttention_32272384262667
// MI455X (gfx1250) — matrix operand order repaired and checked by reading, not compiled, not run on hardware
//
#include <hip/hip_runtime.h>

// MultiScaleAttention forward for MI455X (gfx1250, wave32).
// All 1x1-conv / projection GEMMs run on v_wmma_f32_16x16x32_f16 (f32 accum),
// staged through LDS with 16-byte vector ds ops. Non-local attention
// uses wave32 shuffles; BN-stat uses an LDS tree reduction.

typedef __attribute__((ext_vector_type(16))) _Float16 v16h;
typedef __attribute__((ext_vector_type(8)))  _Float16 v8h;
typedef __attribute__((ext_vector_type(8)))  float    v8f;

#define TOTCOLS 98304   // B*N*K = 8*1024*12
#define PTS     8192    // B*N
#define NPTS    1024
#define BATCH   8
#define KNN     12

// ---------------------------------------------------------------- kNN (k=12)
__global__ __launch_bounds__(256)
void knn_kernel(const float* __restrict__ x, int* __restrict__ idx) {
  int t = blockIdx.x * 256 + threadIdx.x;           // 0..8191 = b*1024+n
  if (t >= PTS) return;
  int b = t >> 10, n = t & 1023;
  const float* xb = x + b * 3 * NPTS;
  float x0 = xb[n], x1 = xb[NPTS + n], x2 = xb[2 * NPTS + n];
  float xxn = x0 * x0 + x1 * x1 + x2 * x2;
  float best[KNN]; int bi[KNN];
  #pragma unroll
  for (int j = 0; j < KNN; ++j) { best[j] = -3.0e38f; bi[j] = 0; }
  for (int m = 0; m < NPTS; ++m) {
    float y0 = xb[m], y1 = xb[NPTS + m], y2 = xb[2 * NPTS + m];
    float dot = x0 * y0 + x1 * y1 + x2 * y2;
    float xxm = y0 * y0 + y1 * y1 + y2 * y2;
    float neg = 2.f * dot - xxn - xxm;
    if (neg > best[KNN - 1]) {
      int p = KNN - 1;
      while (p > 0 && neg > best[p - 1]) { best[p] = best[p - 1]; bi[p] = bi[p - 1]; --p; }
      best[p] = neg; bi[p] = m;
    }
  }
  #pragma unroll
  for (int j = 0; j < KNN; ++j) idx[t * KNN + j] = bi[j];
}

// -------------------------------------------- edge features: (6, B*N*K) f32
__global__ __launch_bounds__(256)
void build_h0_kernel(const float* __restrict__ x, const int* __restrict__ idx,
                     float* __restrict__ h0) {
  int e = blockIdx.x * 256 + threadIdx.x;           // = (b*1024+n)*12 + k
  if (e >= TOTCOLS) return;
  int t = e / KNN;
  int b = t >> 10, n = t & 1023;
  int j = idx[e];
  #pragma unroll
  for (int c = 0; c < 3; ++c) {
    h0[(size_t)c * TOTCOLS + e]       = x[b * 3 * NPTS + c * NPTS + j];  // neighbor
    h0[(size_t)(c + 3) * TOTCOLS + e] = x[b * 3 * NPTS + c * NPTS + n];  // center
  }
}

// ------------------------------------------------------------- WMMA GEMM
// D[M x N] = A[M x K] * B[K x N] (+bias per row)(+res), fp32 in/out, f16 MACs.
// Block: 128 thr = 4 waves; tile 32(M) x 128(N); K stepped by 32 via LDS.
// REQUIRES: M % 32 == 0, N % 128 == 0 (all call sites satisfy this).
// A fragment K map: 16*(v/4)+8*half+2*(v%4) -> two contiguous 8-half runs.
// B fragment K map: same as A -> two contiguous 8-half runs; B tile is
// stored transposed ([n][k]) in LDS so every fragment is 2x b128 ds loads.
#define LDA 40   // halfs per LDS row (80B = 20 banks: conflict-spread, 16B-aligned)
#define LDB 40

__global__ __launch_bounds__(128)
void gemm_wmma_kernel(const float* __restrict__ A, const float* __restrict__ Bm,
                      const float* __restrict__ bias, const float* res,
                      float* D, int M, int K, int N, int ldb, int ldd) {
  __shared__ __align__(16) _Float16 lA[32 * LDA];
  __shared__ __align__(16) _Float16 lB[128 * LDB];
  const int tid  = threadIdx.x;
  const int lane = tid & 31;
  const int wv   = tid >> 5;
  const int m0   = blockIdx.y * 32;
  const int n0   = blockIdx.x * 128;
  const int mrow = lane & 15;
  const int half = lane >> 4;

  v8f acc00 = {}, acc01 = {}, acc10 = {}, acc11 = {};

  const int am = tid >> 2;                       // 0..31  (A row)
  const int ak = (tid & 3) * 8;                  // 0,8,16,24 (A k-run)
  const float* Aptr = A + (size_t)(m0 + am) * K;
  const float* Bptr = Bm + n0 + tid;             // this thread's column

  for (int k0 = 0; k0 < K; k0 += 32) {
    bool full = (k0 + 32 <= K);
    // ---- A tile 32x32 -> lA[m][k]; one 16B ds store per thread
    v8h pa;
    if (full) {
      #pragma unroll
      for (int j = 0; j < 8; ++j) pa[j] = (_Float16)Aptr[k0 + ak + j];
    } else {
      #pragma unroll
      for (int j = 0; j < 8; ++j) {
        int kg = k0 + ak + j;
        pa[j] = (_Float16)((kg < K) ? Aptr[kg] : 0.f);
      }
    }
    *(v8h*)(lA + am * LDA + ak) = pa;

    // ---- B tile 32x128 -> lB[n][k] transposed; 4x 16B ds stores per thread
    if (full) {
      #pragma unroll
      for (int jj = 0; jj < 4; ++jj) {
        v8h pb;
        #pragma unroll
        for (int j = 0; j < 8; ++j)
          pb[j] = (_Float16)Bptr[(size_t)(k0 + jj * 8 + j) * ldb];
        *(v8h*)(lB + tid * LDB + jj * 8) = pb;
      }
      if (k0 + 32 < K)                               // global_prefetch_b8
        __builtin_prefetch(Bptr + (size_t)(k0 + 32) * ldb, 0, 1);
    } else {
      #pragma unroll
      for (int jj = 0; jj < 4; ++jj) {
        v8h pb;
        #pragma unroll
        for (int j = 0; j < 8; ++j) {
          int kg = k0 + jj * 8 + j;
          pb[j] = (_Float16)((kg < K) ? Bptr[(size_t)kg * ldb] : 0.f);
        }
        *(v8h*)(lB + tid * LDB + jj * 8) = pb;
      }
    }
    __syncthreads();

    union Frag { v16h v; v8h h[2]; } a0, a1, b0, b1;
    a0.h[0] = *(const v8h*)(lA + mrow * LDA + half * 8);
    a0.h[1] = *(const v8h*)(lA + mrow * LDA + 16 + half * 8);
    a1.h[0] = *(const v8h*)(lA + (16 + mrow) * LDA + half * 8);
    a1.h[1] = *(const v8h*)(lA + (16 + mrow) * LDA + 16 + half * 8);
    const int nn = wv * 32 + mrow;
    b0.h[0] = *(const v8h*)(lB + nn * LDB + half * 8);
    b0.h[1] = *(const v8h*)(lB + nn * LDB + 16 + half * 8);
    b1.h[0] = *(const v8h*)(lB + (nn + 16) * LDB + half * 8);
    b1.h[1] = *(const v8h*)(lB + (nn + 16) * LDB + 16 + half * 8);

    acc00 = __builtin_amdgcn_wmma_f32_16x16x32_f16(false, a0.v, false, b0.v,
                                                   (short)0, acc00, false, false);
    acc01 = __builtin_amdgcn_wmma_f32_16x16x32_f16(false, a0.v, false, b1.v,
                                                   (short)0, acc01, false, false);
    acc10 = __builtin_amdgcn_wmma_f32_16x16x32_f16(false, a1.v, false, b0.v,
                                                   (short)0, acc10, false, false);
    acc11 = __builtin_amdgcn_wmma_f32_16x16x32_f16(false, a1.v, false, b1.v,
                                                   (short)0, acc11, false, false);
    __syncthreads();
  }

  #pragma unroll
  for (int mt = 0; mt < 2; ++mt) {
    v8f aL = mt ? acc10 : acc00;
    v8f aR = mt ? acc11 : acc01;
    #pragma unroll
    for (int s = 0; s < 2; ++s) {
      v8f acc = s ? aR : aL;
      int col = n0 + wv * 32 + s * 16 + mrow;
      #pragma unroll
      for (int r = 0; r < 8; ++r) {
        int m = m0 + mt * 16 + (half << 3) + r;
        float vv = acc[r];
        if (bias) vv += bias[m];
        if (res)  vv += res[(size_t)m * ldd + col];
        D[(size_t)m * ldd + col] = vv;
      }
    }
  }
}

// ----------------------------------------------------- batch-stat BN + ReLU
__global__ __launch_bounds__(256)
void colstats_kernel(const float* __restrict__ X, float* __restrict__ mean,
                     float* __restrict__ var, int ncols) {
  __shared__ float ssum[256], ssq[256];
  int row = blockIdx.x;
  const float* p = X + (size_t)row * ncols;
  float s = 0.f, q = 0.f;
  for (int i = threadIdx.x; i < ncols; i += 256) { float v = p[i]; s += v; q += v * v; }
  ssum[threadIdx.x] = s; ssq[threadIdx.x] = q;
  __syncthreads();
  for (int o = 128; o > 0; o >>= 1) {
    if (threadIdx.x < o) { ssum[threadIdx.x] += ssum[threadIdx.x + o];
                           ssq[threadIdx.x]  += ssq[threadIdx.x + o]; }
    __syncthreads();
  }
  if (threadIdx.x == 0) {
    float m = ssum[0] / ncols;
    mean[row] = m;
    var[row]  = ssq[0] / ncols - m * m;
  }
}

__global__ __launch_bounds__(256)
void bn_relu_kernel(float* X, const float* __restrict__ mean, const float* __restrict__ var,
                    const float* __restrict__ g, const float* __restrict__ b,
                    int ncols, int relu) {
  int row = blockIdx.y;
  int i = blockIdx.x * 256 + threadIdx.x;
  if (i >= ncols) return;
  float inv = rsqrtf(var[row] + 1e-5f);
  float v = (X[(size_t)row * ncols + i] - mean[row]) * inv * g[row] + b[row];
  if (relu) v = fmaxf(v, 0.f);
  X[(size_t)row * ncols + i] = v;
}

// ------------------------------------- aggregate: per-point 12x12 attention
__global__ __launch_bounds__(256)
void agg_attn_kernel(const float* __restrict__ q, const float* __restrict__ k,
                     const float* __restrict__ v, float* __restrict__ o,
                     int C, int npts, int ld) {
  __shared__ float att[8][144];
  int wv = threadIdx.x >> 5, lane = threadIdx.x & 31;
  int pt = blockIdx.x * 8 + wv;
  bool ok = pt < npts;
  int col0 = pt * KNN;
  float scale = rsqrtf((float)C);
  for (int pr = lane; pr < 144; pr += 32) {
    int kk = pr / KNN, ll = pr - kk * KNN;
    float s = 0.f;
    if (ok) for (int c = 0; c < C; ++c)
      s += q[(size_t)c * ld + col0 + kk] * k[(size_t)c * ld + col0 + ll];
    att[wv][pr] = s * scale;
  }
  __syncthreads();
  if (lane < KNN) {                                  // softmax over last axis
    float mx = -3.0e38f;
    for (int l = 0; l < KNN; ++l) mx = fmaxf(mx, att[wv][lane * KNN + l]);
    float sm = 0.f;
    for (int l = 0; l < KNN; ++l) { float e = __expf(att[wv][lane * KNN + l] - mx);
                                    att[wv][lane * KNN + l] = e; sm += e; }
    float inv = 1.f / sm;
    for (int l = 0; l < KNN; ++l) att[wv][lane * KNN + l] *= inv;
  }
  __syncthreads();
  if (ok) for (int t = lane; t < C * KNN; t += 32) {
    int c = t / KNN, kk = t - c * KNN;
    float a = 0.f;
    for (int l = 0; l < KNN; ++l) a += att[wv][kk * KNN + l] * v[(size_t)c * ld + col0 + l];
    o[(size_t)c * ld + col0 + kk] = a;
  }
}

// ---------------------------------- non-local attention (one wave per row o)
__global__ __launch_bounds__(256)
void nl_attn_kernel(const float* __restrict__ Q, const float* __restrict__ Km,
                    const float* __restrict__ V, float* __restrict__ msg,
                    int C, int heads) {
  __shared__ float qs[8][64];
  int wv = threadIdx.x >> 5, lane = threadIdx.x & 31;
  int gw = blockIdx.x * 8 + wv;                      // (b, h, o) flat
  int o  = gw & (NPTS - 1);
  int bh = gw >> 10;
  int h  = bh % heads;
  int b  = bh / heads;
  int hd = C / heads;
  float scale = rsqrtf((float)hd);
  const size_t ld = PTS;
  size_t base = (size_t)(h * hd) * ld + (size_t)b * NPTS;
  for (int d = lane; d < hd; d += 32) qs[wv][d] = Q[base + (size_t)d * ld + o];
  __syncthreads();
  float sc[32];
  #pragma unroll
  for (int j = 0; j < 32; ++j) {                     // scores vs all 1024 keys
    int i = lane + j * 32;
    float s = 0.f;
    for (int d = 0; d < hd; ++d) s += qs[wv][d] * Km[base + (size_t)d * ld + i];
    sc[j] = s * scale;
  }
  float mx = -3.0e38f;
  #pragma unroll
  for (int j = 0; j < 32; ++j) mx = fmaxf(mx, sc[j]);
  for (int ofs = 16; ofs > 0; ofs >>= 1) mx = fmaxf(mx, __shfl_xor(mx, ofs, 32));
  float sm = 0.f;
  #pragma unroll
  for (int j = 0; j < 32; ++j) { sc[j] = __expf(sc[j] - mx); sm += sc[j]; }
  for (int ofs = 16; ofs > 0; ofs >>= 1) sm += __shfl_xor(sm, ofs, 32);
  float inv = 1.f / sm;
  #pragma unroll
  for (int j = 0; j < 32; ++j) sc[j] *= inv;
  for (int d = 0; d < hd; ++d) {                     // msg = w @ V^T
    float a = 0.f;
    #pragma unroll
    for (int j = 0; j < 32; ++j) a += sc[j] * V[base + (size_t)d * ld + lane + j * 32];
    for (int ofs = 16; ofs > 0; ofs >>= 1) a += __shfl_xor(a, ofs, 32);
    if (lane == 0) msg[base + (size_t)d * ld + o] = a;
  }
}

// ------------------------------------------------------------ small helpers
__global__ __launch_bounds__(256)
void maxpool_kernel(const float* __restrict__ H, float* __restrict__ F, int C) {
  int i = blockIdx.x * 256 + threadIdx.x;
  if (i >= C * PTS) return;
  int c = i >> 13, p = i & (PTS - 1);
  const float* s = H + (size_t)c * TOTCOLS + (size_t)p * KNN;
  float m = s[0];
  #pragma unroll
  for (int kk = 1; kk < KNN; ++kk) m = fmaxf(m, s[kk]);
  F[i] = m;
}

__global__ __launch_bounds__(256)
void sub_kernel(const float* __restrict__ F, float* M_, int nelem) {
  int i = blockIdx.x * 256 + threadIdx.x;
  if (i < nelem) M_[i] = F[i] - M_[i];
}

__global__ __launch_bounds__(256)
void rowmean_kernel(const float* __restrict__ X, float* __restrict__ s, int C) {
  __shared__ float red[256];
  int b = blockIdx.x / C, c = blockIdx.x % C;
  const float* p = X + (size_t)c * PTS + (size_t)b * NPTS;
  float a = 0.f;
  for (int n = threadIdx.x; n < NPTS; n += 256) a += p[n];
  red[threadIdx.x] = a;
  __syncthreads();
  for (int o = 128; o > 0; o >>= 1) {
    if (threadIdx.x < o) red[threadIdx.x] += red[threadIdx.x + o];
    __syncthreads();
  }
  if (threadIdx.x == 0) s[blockIdx.x] = red[0] / NPTS;      // s[b*C+c]
}

__global__ __launch_bounds__(256)
void fcgate_kernel(const float* __restrict__ s, const float* __restrict__ Ws1,
                   const float* __restrict__ bs1, const float* __restrict__ Ws2,
                   const float* __restrict__ bs2, float* __restrict__ gate, int C) {
  __shared__ float sv[256];
  __shared__ float hid[16];
  int b = blockIdx.x, hdim = C / 16;
  if ((int)threadIdx.x < C) sv[threadIdx.x] = s[b * C + threadIdx.x];
  __syncthreads();
  if ((int)threadIdx.x < hdim) {
    float a = bs1[threadIdx.x];
    for (int c = 0; c < C; ++c) a += Ws1[threadIdx.x * C + c] * sv[c];
    hid[threadIdx.x] = fmaxf(a, 0.f);
  }
  __syncthreads();
  if ((int)threadIdx.x < C) {
    float a = bs2[threadIdx.x];
    for (int j = 0; j < hdim; ++j) a += Ws2[threadIdx.x * hdim + j] * hid[j];
    gate[b * C + threadIdx.x] = 1.f / (1.f + __expf(-a));
  }
}

__global__ __launch_bounds__(256)
void resgate_kernel(float* F, const float* __restrict__ M3,
                    const float* __restrict__ gate, int C) {
  int i = blockIdx.x * 256 + threadIdx.x;
  if (i >= C * PTS) return;
  int c = i >> 13, col = i & (PTS - 1), b = col >> 10;
  F[i] += M3[i] * gate[b * C + c];
}

__global__ __launch_bounds__(256)
void transpose_out_kernel(const float* __restrict__ E, float* __restrict__ out) {
  int i = blockIdx.x * 256 + threadIdx.x;
  if (i >= BATCH * 256 * NPTS) return;
  int b = i >> 18, r = i & 262143, c = r >> 10, n = r & 1023;
  out[i] = E[(size_t)c * PTS + (size_t)b * NPTS + n];
}

// =========================================================== host orchestration
struct AggP { const float *Wq, *bq, *Wk, *bk, *Wv, *bv, *Wp, *bp; };
struct NLP  { const float *Wq, *bq, *Wk, *bk, *Wv, *bv, *W1, *b1, *g1, *be1,
                          *W2, *b2, *g2, *be2, *W3, *b3, *Ws1, *bs1, *Ws2, *bs2; };

extern "C" void kernel_launch(void* const* d_in, const int* in_sizes, int n_in,
                              void* d_out, int out_size, void* d_ws, size_t ws_size,
                              hipStream_t stream) {
  (void)in_sizes; (void)n_in; (void)out_size; (void)ws_size;

  // ---- inputs in setup_inputs() dict (insertion) order ----
  int q = 0;
  auto F = [&](int i) { return (const float*)d_in[i]; };
  const float* X   = F(q++);
  const float* cW1 = F(q++); const float* cg1 = F(q++); const float* cbe1 = F(q++);
  const float* cW2 = F(q++); const float* cg2 = F(q++); const float* cbe2 = F(q++);
  const float* cW3 = F(q++); const float* cg3 = F(q++); const float* cbe3 = F(q++);
  const float* cW4 = F(q++); const float* cg4 = F(q++); const float* cbe4 = F(q++);
  AggP agg[3];
  for (int i = 0; i < 3; ++i) {
    agg[i].Wq = F(q++); agg[i].bq = F(q++); agg[i].Wk = F(q++); agg[i].bk = F(q++);
    agg[i].Wv = F(q++); agg[i].bv = F(q++); agg[i].Wp = F(q++); agg[i].bp = F(q++);
  }
  NLP att[6];   // att1, att2, att3, att4[0..2]
  for (int i = 0; i < 6; ++i) {
    att[i].Wq = F(q++); att[i].bq = F(q++); att[i].Wk = F(q++); att[i].bk = F(q++);
    att[i].Wv = F(q++); att[i].bv = F(q++); att[i].W1 = F(q++); att[i].b1 = F(q++);
    att[i].g1 = F(q++); att[i].be1 = F(q++); att[i].W2 = F(q++); att[i].b2 = F(q++);
    att[i].g2 = F(q++); att[i].be2 = F(q++); att[i].W3 = F(q++); att[i].b3 = F(q++);
    att[i].Ws1 = F(q++); att[i].bs1 = F(q++); att[i].Ws2 = F(q++); att[i].bs2 = F(q++);
  }

  // ---- workspace bump allocator ----
  size_t off = 0;
  auto alloc = [&](size_t nbytes) -> void* {
    void* p = (char*)d_ws + off;
    off += (nbytes + 255) & ~(size_t)255;
    return p;
  };
  int*   IDX = (int*)alloc((size_t)PTS * KNN * 4);
  float* H0  = (float*)alloc((size_t)6 * TOTCOLS * 4);
  float* HA  = (float*)alloc((size_t)128 * TOTCOLS * 4);
  float* HB  = (float*)alloc((size_t)128 * TOTCOLS * 4);
  const int CHCOLS = NPTS * KNN;                       // 12288 cols / chunk
  float* TQ = (float*)alloc((size_t)128 * CHCOLS * 4);
  float* TK = (float*)alloc((size_t)128 * CHCOLS * 4);
  float* TV = (float*)alloc((size_t)128 * CHCOLS * 4);
  float* TO = (float*)alloc((size_t)128 * CHCOLS * 4);
  float* NQ  = (float*)alloc((size_t)256 * PTS * 4);
  float* NK  = (float*)alloc((size_t)256 * PTS * 4);
  float* NV  = (float*)alloc((size_t)256 * PTS * 4);
  float* NM  = (float*)alloc((size_t)256 * PTS * 4);
  float* NM1 = (float*)alloc((size_t)256 * PTS * 4);
  float* NM2 = (float*)alloc((size_t)256 * PTS * 4);
  float* NM3 = (float*)alloc((size_t)256 * PTS * 4);
  float* CAT = (float*)alloc((size_t)256 * PTS * 4);
  float* E   = (float*)alloc((size_t)256 * PTS * 4);
  float* SVEC  = (float*)alloc(BATCH * 256 * 4);
  float* SGATE = (float*)alloc(BATCH * 256 * 4);
  float* SMEAN = (float*)alloc(256 * 4);
  float* SVAR  = (float*)alloc(256 * 4);

  auto gemm = [&](const float* A, const float* Bm, const float* bias,
                  const float* res, float* D, int M, int K, int N, int ldb, int ldd) {
    dim3 g((N + 127) / 128, M / 32);
    gemm_wmma_kernel<<<g, dim3(128), 0, stream>>>(A, Bm, bias, res, D, M, K, N, ldb, ldd);
  };
  auto bnrelu = [&](float* Xb, const float* g, const float* b, int M, int ncols, int relu) {
    colstats_kernel<<<M, 256, 0, stream>>>(Xb, SMEAN, SVAR, ncols);
    bn_relu_kernel<<<dim3((ncols + 255) / 256, M), 256, 0, stream>>>(Xb, SMEAN, SVAR, g, b, ncols, relu);
  };
  auto aggregate = [&](float* H, int C, const AggP& p) {
    for (int ch = 0; ch < 8; ++ch) {                       // 1024 points / chunk
      const float* Bp = H + ch * CHCOLS;
      gemm(p.Wq, Bp, p.bq, nullptr, TQ, C, C, CHCOLS, TOTCOLS, CHCOLS);
      gemm(p.Wk, Bp, p.bk, nullptr, TK, C, C, CHCOLS, TOTCOLS, CHCOLS);
      gemm(p.Wv, Bp, p.bv, nullptr, TV, C, C, CHCOLS, TOTCOLS, CHCOLS);
      agg_attn_kernel<<<NPTS / 8, 256, 0, stream>>>(TQ, TK, TV, TO, C, NPTS, CHCOLS);
      gemm(p.Wp, TO, p.bp, H + ch * CHCOLS, H + ch * CHCOLS, C, C, CHCOLS, CHCOLS, TOTCOLS);
    }
  };
  auto nonlocal = [&](float* Ft, int C, const NLP& p) {
    int h2 = C / 2;
    gemm(p.Wq, Ft, p.bq, nullptr, NQ, C, C, PTS, PTS, PTS);
    gemm(p.Wk, Ft, p.bk, nullptr, NK, C, C, PTS, PTS, PTS);
    gemm(p.Wv, Ft, p.bv, nullptr, NV, C, C, PTS, PTS, PTS);
    nl_attn_kernel<<<BATCH * 4 * NPTS / 8, 256, 0, stream>>>(NQ, NK, NV, NM, C, 4);
    sub_kernel<<<(C * PTS + 255) / 256, 256, 0, stream>>>(Ft, NM, C * PTS);   // m = feat - msg
    gemm(p.W1, NM, p.b1, nullptr, NM1, h2, C, PTS, PTS, PTS);
    bnrelu(NM1, p.g1, p.be1, h2, PTS, 1);
    gemm(p.W2, NM1, p.b2, nullptr, NM2, h2, h2, PTS, PTS, PTS);
    bnrelu(NM2, p.g2, p.be2, h2, PTS, 1);
    gemm(p.W3, NM2, p.b3, nullptr, NM3, C, h2, PTS, PTS, PTS);
    rowmean_kernel<<<BATCH * C, 256, 0, stream>>>(NM3, SVEC, C);
    fcgate_kernel<<<BATCH, 256, 0, stream>>>(SVEC, p.Ws1, p.bs1, p.Ws2, p.bs2, SGATE, C);
    resgate_kernel<<<(C * PTS + 255) / 256, 256, 0, stream>>>(Ft, NM3, SGATE, C);
  };

  // ---------------- forward ----------------
  knn_kernel<<<PTS / 256, 256, 0, stream>>>(X, IDX);
  build_h0_kernel<<<(TOTCOLS + 255) / 256, 256, 0, stream>>>(X, IDX, H0);

  // stage 1: conv W1 (64x6) -> BN+ReLU -> aggregate -> maxpool -> nonlocal
  gemm(cW1, H0, nullptr, nullptr, HA, 64, 6, TOTCOLS, TOTCOLS, TOTCOLS);
  bnrelu(HA, cg1, cbe1, 64, TOTCOLS, 1);
  aggregate(HA, 64, agg[0]);
  maxpool_kernel<<<(64 * PTS + 255) / 256, 256, 0, stream>>>(HA, CAT, 64);
  nonlocal(CAT, 64, att[0]);

  // stage 2: conv W2 (64x64)
  gemm(cW2, HA, nullptr, nullptr, HB, 64, 64, TOTCOLS, TOTCOLS, TOTCOLS);
  bnrelu(HB, cg2, cbe2, 64, TOTCOLS, 1);
  aggregate(HB, 64, agg[1]);
  maxpool_kernel<<<(64 * PTS + 255) / 256, 256, 0, stream>>>(HB, CAT + (size_t)64 * PTS, 64);
  nonlocal(CAT + (size_t)64 * PTS, 64, att[1]);

  // stage 3: conv W3 (128x64)
  gemm(cW3, HB, nullptr, nullptr, HA, 128, 64, TOTCOLS, TOTCOLS, TOTCOLS);
  bnrelu(HA, cg3, cbe3, 128, TOTCOLS, 1);
  aggregate(HA, 128, agg[2]);
  maxpool_kernel<<<(128 * PTS + 255) / 256, 256, 0, stream>>>(HA, CAT + (size_t)128 * PTS, 128);
  nonlocal(CAT + (size_t)128 * PTS, 128, att[2]);

  // fuse: conv W4 (256x256) + BN + ReLU, then 3 non-local blocks
  gemm(cW4, CAT, nullptr, nullptr, E, 256, 256, PTS, PTS, PTS);
  bnrelu(E, cg4, cbe4, 256, PTS, 1);
  nonlocal(E, 256, att[3]);
  nonlocal(E, 256, att[4]);
  nonlocal(E, 256, att[5]);

  transpose_out_kernel<<<(BATCH * 256 * NPTS + 255) / 256, 256, 0, stream>>>(E, (float*)d_out);
}
